// MockAWQQwen3_5MoE_39874476376663
// MI455X (gfx1250) — matrix operand order repaired and checked by reading, not compiled, not run on hardware
//
#include <hip/hip_runtime.h>

// ---------------- problem constants ----------------
constexpr int kE    = 64;    // experts
constexpr int kTopK = 8;
constexpr int kI    = 512;   // intermediate
constexpr int kH    = 1024;  // hidden
constexpr int kT    = 2048;  // tokens

constexpr int kEI    = kE * kI;      // 32768
constexpr int kEH    = kE * kH;      // 65536
constexpr int kNint1 = kEI / 8;      // 4096 int32 cols of qweight1
constexpr int kNint2 = kEH / 8;      // 8192 int32 cols of qweight2

// LDS strides (halves); pad keeps 16B alignment (stride*2 % 16 == 0) and breaks
// power-of-two bank alignment
constexpr int KS1 = kH + 8;          // 1032 (2064 B rows)
constexpr int KS2 = kI + 8;          // 520  (1040 B rows)

typedef unsigned short u16;
typedef __attribute__((ext_vector_type(8)))  float v8f;
typedef __attribute__((ext_vector_type(16))) __bf16 v16bf;
typedef __attribute__((ext_vector_type(16))) u16    v16u;

struct alignas(16) U4 { unsigned int x, y, z, w; };  // 16B chunk -> b128 ops
union Frag { v16u u; U4 q[2]; };

__device__ __forceinline__ u16 f2bf(float f) {   // f32 -> bf16 RNE
  unsigned int u = __float_as_uint(f);
  unsigned int r = u + 0x7FFFu + ((u >> 16) & 1u);
  return (u16)(r >> 16);
}

// CDNA5 async global->LDS copy (VGPR-bypassing, ASYNCcnt-tracked).
// lds_off: byte offset from wave LDS base; gptr: global source (16B aligned).
__device__ __forceinline__ void async_copy16(unsigned lds_off, const void* gptr) {
  asm volatile("global_load_async_to_lds_b128 %0, %1, off"
               :: "v"(lds_off), "v"((unsigned long long)(uintptr_t)gptr)
               : "memory");
}
__device__ __forceinline__ void wait_async0() {
#if __has_builtin(__builtin_amdgcn_s_wait_asynccnt)
  __builtin_amdgcn_s_wait_asynccnt(0);
#else
  asm volatile("s_wait_asynccnt 0" ::: "memory");
#endif
}

// AWQ nibble order: dense col (pack*8 + c) uses shift kShift[c]
__constant__ int kShift[8] = {0, 16, 4, 20, 8, 24, 12, 28};

// ---------------- x -> bf16 (4-wide) ----------------
__global__ void cvt_kernel(const float* __restrict__ x,
                           unsigned long long* __restrict__ xb4, int n4) {
  int i = blockIdx.x * blockDim.x + threadIdx.x;
  if (i < n4) {
    float4 v = ((const float4*)x)[i];
    unsigned long long lo = (unsigned)f2bf(v.x) | ((unsigned)f2bf(v.y) << 16);
    unsigned long long hi = (unsigned)f2bf(v.z) | ((unsigned)f2bf(v.w) << 16);
    xb4[i] = lo | (hi << 32);
  }
}

// ---------------- routing: softmax + top8 + expert lists ----------------
__global__ __launch_bounds__(kE) void route_kernel(
    const float* __restrict__ x, const float* __restrict__ gw,
    int* __restrict__ cnt, int* __restrict__ slots, float* __restrict__ wts) {
  __shared__ float sp[kE];
  __shared__ float smx;
  int t = blockIdx.x, e = threadIdx.x;
  const float4* xr = (const float4*)(x + (size_t)t * kH);
  const float4* gr = (const float4*)(gw + (size_t)e * kH);
  float acc = 0.f;
  for (int i = 0; i < kH / 4; ++i) {
    float4 a = xr[i], b = gr[i];
    acc += a.x * b.x + a.y * b.y + a.z * b.z + a.w * b.w;
  }
  sp[e] = acc;
  __syncthreads();
  if (e == 0) {
    float m = sp[0];
    for (int i = 1; i < kE; ++i) m = fmaxf(m, sp[i]);
    smx = m;
  }
  __syncthreads();
  sp[e] = __expf(sp[e] - smx);       // unnormalized probs (norm cancels in renorm)
  __syncthreads();
  if (e == 0) {
    float tv[kTopK]; int ti[kTopK];
    for (int k = 0; k < kTopK; ++k) {
      float best = -1.f; int bi = 0;
      for (int i = 0; i < kE; ++i)
        if (sp[i] > best) { best = sp[i]; bi = i; }
      tv[k] = best; ti[k] = bi; sp[bi] = -1.f;
    }
    float s = 0.f;
    for (int k = 0; k < kTopK; ++k) s += tv[k];
    float inv = 1.f / s;
    for (int k = 0; k < kTopK; ++k) {
      int pos = atomicAdd(&cnt[ti[k]], 1);
      slots[ti[k] * kT + pos] = t * kTopK + k;
      wts  [ti[k] * kT + pos] = tv[k] * inv;
    }
  }
}

// ---------------- GEMM1: h[slot] = bf16( silu(x @ W1[e].T) * cw ) ----------------
// grid (kE, kI/64), block 256 (8 waves); LDS: Bs[64][KS1] + As[64][KS1] = 264KB
// wave w: cols (w&3)*16, rows (w>>2)*32 (2 accumulators share one B fragment)
__global__ __launch_bounds__(256) void gemm1_kernel(
    const u16* __restrict__ xb,
    const int* __restrict__ qw1, const int* __restrict__ qz1,
    const float* __restrict__ sc1,
    const int* __restrict__ cnt, const int* __restrict__ slots,
    const float* __restrict__ wts, u16* __restrict__ hbuf) {
  extern __shared__ char smem[];
  u16* Bs = (u16*)smem;                          // 64 * KS1 halves
  u16* As = (u16*)smem + 64 * KS1;               // 64 * KS1 halves
  constexpr unsigned kAsByteOff = 64u * KS1 * 2u;

  const int e  = blockIdx.x;
  const int nb = blockIdx.y;                     // 64-col tile of I
  const int N0 = e * kI + nb * 64;               // dense col base in (K, E*I)
  const int p0 = N0 >> 3;                        // int32 pack base
  const int tid = threadIdx.x;

  // dequant W1 tile into LDS once (each weight element dequantized exactly once)
  for (int idx = tid; idx < kH * 8; idx += 256) {
    int k = idx >> 3, pp = idx & 7;
    int g = k >> 7;                              // k / 128
    unsigned qw = (unsigned)qw1[(size_t)k * kNint1 + p0 + pp];
    unsigned qz = (unsigned)qz1[(size_t)g * kNint1 + p0 + pp];
    const float* srow = sc1 + (size_t)g * kEI + N0 + pp * 8;
    if (k + 256 < kH)
      __builtin_prefetch(qw1 + (size_t)(k + 256) * kNint1 + p0 + pp, 0, 1);
    int colb = pp * 8;
#pragma unroll
    for (int c = 0; c < 8; ++c) {
      int q = (int)((qw >> kShift[c]) & 0xF);
      int z = (int)((qz >> kShift[c]) & 0xF);
      Bs[(colb + c) * KS1 + k] = f2bf((float)(q - z) * srow[c]);
    }
  }

  const int cntE = cnt[e];
  const int wave = tid >> 5, lane = tid & 31;
  const int half = lane >> 4, l15 = lane & 15;
  const int colw = (wave & 3) * 16;              // wave's 16-col sub-tile
  const int mh   = (wave >> 2) * 32;             // wave's 32-row sub-tile

  for (int mt = 0; mt * 64 < cntE; ++mt) {
    __syncthreads();
    // async-stage up to 64 gathered x rows (bf16) into LDS (16B chunks).
    // Tail rows are left stale: they only feed output rows that are discarded.
    for (int idx = tid; idx < 64 * (kH / 8); idx += 256) {
      int r = idx >> 7, c = idx & 127;
      int m = mt * 64 + r;
      if (m < cntE) {
        int t = slots[e * kT + m] >> 3;          // slot / kTopK
        async_copy16(kAsByteOff + (unsigned)(r * KS1 + c * 8) * 2u,
                     xb + (size_t)t * kH + c * 8);
      }
    }
    wait_async0();
    __syncthreads();

    v8f acc0 = {}, acc1 = {};
    for (int kk = 0; kk < kH; kk += 32) {
      Frag fb;
      const u16* bp = Bs + (colw + l15) * KS1 + kk + half * 8;   // B: N=l15
      fb.q[0] = *(const U4*)(bp);
      fb.q[1] = *(const U4*)(bp + 16);
      Frag fa0, fa1;
      const u16* a0 = As + (mh + l15) * KS1 + kk + half * 8;     // A: M=l15
      const u16* a1 = a0 + 16 * KS1;
      fa0.q[0] = *(const U4*)(a0);
      fa0.q[1] = *(const U4*)(a0 + 16);
      fa1.q[0] = *(const U4*)(a1);
      fa1.q[1] = *(const U4*)(a1 + 16);
      acc0 = __builtin_amdgcn_wmma_f32_16x16x32_bf16(
          false, __builtin_bit_cast(v16bf, fa0.u),
          false, __builtin_bit_cast(v16bf, fb.u),
          (short)0, acc0, false, false);
      acc1 = __builtin_amdgcn_wmma_f32_16x16x32_bf16(
          false, __builtin_bit_cast(v16bf, fa1.u),
          false, __builtin_bit_cast(v16bf, fb.u),
          (short)0, acc1, false, false);
    }

    // D layout: vgpr i -> M = i + 8*half, N = l15
#pragma unroll
    for (int i = 0; i < 8; ++i) {
      int m0 = mt * 64 + mh + i + half * 8;
      int m1 = m0 + 16;
      int col = nb * 64 + colw + l15;
      if (m0 < cntE) {
        int slot = slots[e * kT + m0];
        float wt = wts[e * kT + m0];
        float v = acc0[i];
        hbuf[(size_t)slot * kI + col] = f2bf(v / (1.f + __expf(-v)) * wt);
      }
      if (m1 < cntE) {
        int slot = slots[e * kT + m1];
        float wt = wts[e * kT + m1];
        float v = acc1[i];
        hbuf[(size_t)slot * kI + col] = f2bf(v / (1.f + __expf(-v)) * wt);
      }
    }
  }
}

// ---------------- GEMM2: out[t] += h[slot] @ W2[e].T (atomic f32) ----------------
// grid (kE, kH/64), block 256; LDS: Bs[64][KS2] + As[64][KS2] = 133KB
__global__ __launch_bounds__(256) void gemm2_kernel(
    const u16* __restrict__ hbuf,
    const int* __restrict__ qw2, const int* __restrict__ qz2,
    const float* __restrict__ sc2,
    const int* __restrict__ cnt, const int* __restrict__ slots,
    float* __restrict__ out) {
  extern __shared__ char smem[];
  u16* Bs = (u16*)smem;                          // 64 * KS2
  u16* As = (u16*)smem + 64 * KS2;               // 64 * KS2
  constexpr unsigned kAsByteOff = 64u * KS2 * 2u;

  const int e  = blockIdx.x;
  const int nb = blockIdx.y;                     // 64-col tile of H
  const int N0 = e * kH + nb * 64;
  const int p0 = N0 >> 3;
  const int tid = threadIdx.x;

  for (int idx = tid; idx < kI * 8; idx += 256) {
    int k = idx >> 3, pp = idx & 7;
    int g = k >> 7;
    unsigned qw = (unsigned)qw2[(size_t)k * kNint2 + p0 + pp];
    unsigned qz = (unsigned)qz2[(size_t)g * kNint2 + p0 + pp];
    const float* srow = sc2 + (size_t)g * kEH + N0 + pp * 8;
    if (k + 256 < kI)
      __builtin_prefetch(qw2 + (size_t)(k + 256) * kNint2 + p0 + pp, 0, 1);
    int colb = pp * 8;
#pragma unroll
    for (int c = 0; c < 8; ++c) {
      int q = (int)((qw >> kShift[c]) & 0xF);
      int z = (int)((qz >> kShift[c]) & 0xF);
      Bs[(colb + c) * KS2 + k] = f2bf((float)(q - z) * srow[c]);
    }
  }

  const int cntE = cnt[e];
  const int wave = tid >> 5, lane = tid & 31;
  const int half = lane >> 4, l15 = lane & 15;
  const int colw = (wave & 3) * 16;
  const int mh   = (wave >> 2) * 32;

  for (int mt = 0; mt * 64 < cntE; ++mt) {
    __syncthreads();
    for (int idx = tid; idx < 64 * (kI / 8); idx += 256) {
      int r = idx >> 6, c = idx & 63;
      int m = mt * 64 + r;
      if (m < cntE) {
        int slot = slots[e * kT + m];
        async_copy16(kAsByteOff + (unsigned)(r * KS2 + c * 8) * 2u,
                     hbuf + (size_t)slot * kI + c * 8);
      }
    }
    wait_async0();
    __syncthreads();

    v8f acc0 = {}, acc1 = {};
    for (int kk = 0; kk < kI; kk += 32) {
      Frag fb;
      const u16* bp = Bs + (colw + l15) * KS2 + kk + half * 8;
      fb.q[0] = *(const U4*)(bp);
      fb.q[1] = *(const U4*)(bp + 16);
      Frag fa0, fa1;
      const u16* a0 = As + (mh + l15) * KS2 + kk + half * 8;
      const u16* a1 = a0 + 16 * KS2;
      fa0.q[0] = *(const U4*)(a0);
      fa0.q[1] = *(const U4*)(a0 + 16);
      fa1.q[0] = *(const U4*)(a1);
      fa1.q[1] = *(const U4*)(a1 + 16);
      acc0 = __builtin_amdgcn_wmma_f32_16x16x32_bf16(
          false, __builtin_bit_cast(v16bf, fa0.u),
          false, __builtin_bit_cast(v16bf, fb.u),
          (short)0, acc0, false, false);
      acc1 = __builtin_amdgcn_wmma_f32_16x16x32_bf16(
          false, __builtin_bit_cast(v16bf, fa1.u),
          false, __builtin_bit_cast(v16bf, fb.u),
          (short)0, acc1, false, false);
    }

#pragma unroll
    for (int i = 0; i < 8; ++i) {
      int m0 = mt * 64 + mh + i + half * 8;
      int m1 = m0 + 16;
      int col = nb * 64 + colw + l15;
      if (m0 < cntE) {
        int t = slots[e * kT + m0] >> 3;
        atomicAdd(out + (size_t)t * kH + col, acc0[i]);
      }
      if (m1 < cntE) {
        int t = slots[e * kT + m1] >> 3;
        atomicAdd(out + (size_t)t * kH + col, acc1[i]);
      }
    }
  }
}

// ---------------- host ----------------
extern "C" void kernel_launch(void* const* d_in, const int* in_sizes, int n_in,
                              void* d_out, int out_size, void* d_ws, size_t ws_size,
                              hipStream_t stream) {
  const float* x   = (const float*)d_in[0];
  const float* gw  = (const float*)d_in[1];
  const int*   qw1 = (const int*)d_in[2];
  const int*   qz1 = (const int*)d_in[3];
  const float* sc1 = (const float*)d_in[4];
  const int*   qw2 = (const int*)d_in[5];
  const int*   qz2 = (const int*)d_in[6];
  const float* sc2 = (const float*)d_in[7];
  float* out = (float*)d_out;

  // workspace layout
  char* w = (char*)d_ws;
  u16*   xb    = (u16*)w;                                  // 4,194,304 B
  u16*   hbuf  = (u16*)(w + 4194304);                      // 16,777,216 B
  int*   cnt   = (int*)(w + 4194304 + 16777216);           // 256 B (pad 1024)
  int*   slots = (int*)(w + 20971520 + 1024);              // 524,288 B
  float* wts   = (float*)(w + 20971520 + 1024 + 524288);   // 524,288 B

  hipMemsetAsync(cnt, 0, kE * sizeof(int), stream);
  hipMemsetAsync(out, 0, (size_t)out_size * sizeof(float), stream);

  int n4 = (kT * kH) / 4;
  cvt_kernel<<<(n4 + 255) / 256, 256, 0, stream>>>(x, (unsigned long long*)xb, n4);
  route_kernel<<<kT, kE, 0, stream>>>(x, gw, cnt, slots, wts);

  size_t smem1 = (size_t)128 * KS1 * sizeof(u16);          // 264,192 B (<= 320KB/WGP)
  size_t smem2 = (size_t)128 * KS2 * sizeof(u16);          // 133,120 B
  (void)hipFuncSetAttribute((const void*)gemm1_kernel,
      hipFuncAttributeMaxDynamicSharedMemorySize, (int)smem1);
  (void)hipFuncSetAttribute((const void*)gemm2_kernel,
      hipFuncAttributeMaxDynamicSharedMemorySize, (int)smem2);

  gemm1_kernel<<<dim3(kE, kI / 64), 256, smem1, stream>>>(
      xb, qw1, qz1, sc1, cnt, slots, wts, hbuf);
  gemm2_kernel<<<dim3(kE, kH / 64), 256, smem2, stream>>>(
      hbuf, qw2, qz2, sc2, cnt, slots, out);
}
